// NeuralSDEWithJump_69784628626031
// MI455X (gfx1250) — matrix operand order repaired and checked by reading, not compiled, not run on hardware
//
#include <hip/hip_runtime.h>
#include <hip/hip_bf16.h>
#include <math.h>

// ---------------------------------------------------------------------------
// NeuralSDEWithJump for MI455X (gfx1250), wave32 + WMMA bf16 16x16x32.
//   Kernel 1: pack weights -> fragment-major bf16 (WMMA B layout), fuse biases
//   Kernel 2: LSTM encoder, 64 WGs x 32 batch rows, h/c resident, 128 steps
//             (async x-staging, AS1 base+imm weight stream, B double-buffer)
//   Kernel 3: Euler-Maruyama SDE, fused drift|diffusion GEMM + Threefry RNG
// ---------------------------------------------------------------------------

typedef __bf16 bf16;
typedef __attribute__((ext_vector_type(16))) __bf16 v16bf;
typedef __attribute__((ext_vector_type(8)))  float  v8f;
typedef __attribute__((ext_vector_type(2)))  int    v2i;

#define AS1 __attribute__((address_space(1)))
#define AS3 __attribute__((address_space(3)))

#if __has_builtin(__builtin_amdgcn_global_load_async_to_lds_b64) && \
    __has_builtin(__builtin_amdgcn_s_wait_asynccnt)
#define HAS_ASYNC 1
#else
#define HAS_ASYNC 0
#endif

#define B_DIM   2048
#define L_DIM   128
#define I_DIM   64
#define H_DIM   512
#define O_DIM   64
#define T_TOT   256
#define K_LSTM  576     // I + H
#define N_GATE  2048    // 4*H
#define KC_LSTM 18      // 576/32 k-chunks
#define CT_LSTM 128     // 2048/16 col tiles
#define K_SDE   576     // H + O
#define N_SDE   128     // drift(64) | diffusion(64)
#define KC_SDE  18
#define CT_SDE  8

// --- A-matrix (16-bit, 16x32) lane/element mapping (ISA 7.12.2) -------------
__device__ __forceinline__ int a_lane(int ko, int M) {
  int h = (ko < 16) ? (ko >> 3) : ((ko - 16) >> 3);
  return h * 16 + M;
}
__device__ __forceinline__ int a_elem(int ko) {
  return (ko < 16) ? (ko & 7) : (8 + ((ko - 16) & 7));
}

__device__ __forceinline__ v8f wmma_bf16(v16bf a, v16bf b, v8f c) {
  return __builtin_amdgcn_wmma_f32_16x16x32_bf16(false, a, false, b,
                                                 (short)0, c, false, false);
}

__device__ __forceinline__ float sigm(float x) { return 1.f / (1.f + __expf(-x)); }

__device__ __forceinline__ void gprefetch(const AS1 void* p) {
#if __has_builtin(__builtin_amdgcn_global_prefetch)
  __builtin_amdgcn_global_prefetch((AS1 void*)p, 0);
#else
  __builtin_prefetch((const void*)p, 0, 1);
#endif
}

// --- Threefry-2x32-20 counter RNG ------------------------------------------
__device__ __forceinline__ uint2 tf2x32(unsigned k0, unsigned k1,
                                        unsigned c0, unsigned c1) {
  unsigned ks2 = k0 ^ k1 ^ 0x1BD11BDAu;
  unsigned x0 = c0 + k0, x1 = c1 + k1;
#define TFR(r) { x0 += x1; x1 = (x1 << r) | (x1 >> (32 - r)); x1 ^= x0; }
  TFR(13) TFR(15) TFR(26) TFR(6)   x0 += k1;  x1 += ks2 + 1u;
  TFR(17) TFR(29) TFR(16) TFR(24)  x0 += ks2; x1 += k0 + 2u;
  TFR(13) TFR(15) TFR(26) TFR(6)   x0 += k0;  x1 += k1 + 3u;
  TFR(17) TFR(29) TFR(16) TFR(24)  x0 += k1;  x1 += ks2 + 4u;
  TFR(13) TFR(15) TFR(26) TFR(6)   x0 += ks2; x1 += k0 + 5u;
#undef TFR
  uint2 r; r.x = x0; r.y = x1; return r;
}
__device__ __forceinline__ float bm_normal(unsigned w0, unsigned w1) {
  float u1 = (float)(w0 >> 8) * 5.9604645e-8f + 2.9802322e-8f;  // (0,1)
  float u2 = (float)(w1 >> 8) * 5.9604645e-8f;
  float rad = __builtin_sqrtf(-2.f * __logf(u1));
  return rad * __cosf(6.28318530718f * u2);
}

// ---------------------------------------------------------------------------
// Kernel 1: pack weights into WMMA-fragment-major bf16, fuse biases
// B-frag layout (32x16 bf16): lane -> N = lane&15 ; K = e<8 ? (lane>>4)*8+e : 16+(lane>>4)*8+(e-8)
// ---------------------------------------------------------------------------
__global__ __launch_bounds__(256) void pack_kernel(
    const float* __restrict__ W_ih, const float* __restrict__ W_hh,
    const float* __restrict__ b_ih, const float* __restrict__ b_hh,
    const float* __restrict__ drift_W, const float* __restrict__ drift_b,
    const float* __restrict__ diff_W, const float* __restrict__ diff_b,
    bf16* __restrict__ Wp, bf16* __restrict__ Wsp,
    float* __restrict__ biasG, float* __restrict__ bias2) {
  int idx = blockIdx.x * 256 + threadIdx.x;
  const int NW  = KC_LSTM * CT_LSTM * 32 * 16;  // 1179648
  const int NW2 = KC_SDE  * CT_SDE  * 32 * 16;  //   73728
  if (idx < NW) {                               // LSTM weights [K=576][N=2048]
    int e = idx & 15, lane = (idx >> 4) & 31, ct = (idx >> 9) & 127, kc = idx >> 16;
    int K = kc * 32 + ((e < 8) ? ((lane >> 4) * 8 + e) : (16 + (lane >> 4) * 8 + (e - 8)));
    int N = ct * 16 + (lane & 15);
    float v = (K < I_DIM) ? W_ih[N * I_DIM + K] : W_hh[N * H_DIM + (K - I_DIM)];
    Wp[idx] = (bf16)v;
    return;
  }
  idx -= NW;
  if (idx < NW2) {                              // SDE weights [K=576][N=128]
    int e = idx & 15, lane = (idx >> 4) & 31, ct = (idx >> 9) & 7, kc = idx >> 12;
    int K = kc * 32 + ((e < 8) ? ((lane >> 4) * 8 + e) : (16 + (lane >> 4) * 8 + (e - 8)));
    int N = ct * 16 + (lane & 15);
    float v = (N < O_DIM) ? drift_W[N * K_SDE + K] : diff_W[(N - O_DIM) * K_SDE + K];
    Wsp[idx] = (bf16)v;
    return;
  }
  idx -= NW2;
  if (idx < N_GATE) { biasG[idx] = b_ih[idx] + b_hh[idx]; return; }
  idx -= N_GATE;
  if (idx < N_SDE)  { bias2[idx] = (idx < O_DIM) ? drift_b[idx] : diff_b[idx - O_DIM]; }
}

// ---------------------------------------------------------------------------
// Kernel 2: LSTM. 64 WGs x 1024 threads (32 waves). WG owns 32 batch rows.
// ---------------------------------------------------------------------------
__global__ __launch_bounds__(1024) void lstm_kernel(
    const float* __restrict__ xseq, const bf16* __restrict__ Wp,
    const float* __restrict__ biasG, bf16* __restrict__ Hp) {
  __shared__ __align__(128) bf16 Abuf[2 * KC_LSTM * 512];  // 36864 B, A-frag packed
#if HAS_ASYNC
  __shared__ __align__(16) float Xstage[32 * I_DIM];       // 8 KB raw x staging
#endif
  const int tid  = threadIdx.x;
  const int wave = tid >> 5;          // 0..31 : hidden-column tile
  const int lane = tid & 31;
  const int bat0 = blockIdx.x * 32;
  const int jcol = wave * 16 + (lane & 15);   // hidden column in C-frag

  // per-thread gate biases are time-invariant: keep in registers
  const float bias_i = biasG[jcol];
  const float bias_f = biasG[H_DIM + jcol];
  const float bias_g = biasG[2 * H_DIM + jcol];
  const float bias_o = biasG[3 * H_DIM + jcol];

  for (int i = tid; i < 2 * KC_LSTM * 512; i += 1024) Abuf[i] = (bf16)0.f;  // h0 = 0
  __syncthreads();
  // stage x_0 into packed A (K = 0..63)
  for (int q = tid; q < 32 * I_DIM; q += 1024) {
    int r = q >> 6, k = q & 63;
    float x = xseq[((size_t)(bat0 + r) * L_DIM + 0) * I_DIM + k];
    int kc = k >> 5, ko = k & 31, rt = r >> 4, M = r & 15;
    Abuf[(rt * KC_LSTM + kc) * 512 + a_lane(ko, M) * 16 + a_elem(ko)] = (bf16)x;
  }
  __syncthreads();

  float creg[2][8], hnew[2][8];
  #pragma unroll
  for (int rt = 0; rt < 2; ++rt)
    #pragma unroll
    for (int v = 0; v < 8; ++v) creg[rt][v] = 0.f;

  // thread-varying byte offset of this wave's column tile within a k-chunk:
  // fragment (kc,g) lives at  base + voff + kc*131072 + g*32768
  const unsigned voff = (unsigned)(wave * 1024 + lane * 32);

  for (int t = 0; t < L_DIM; ++t) {
#if HAS_ASYNC
    // overlap: async-stage x_{t+1} (raw f32) into LDS during the GEMM phase
    if (t + 1 < L_DIM) {
      int r = tid >> 5, k = (tid * 2) & 63;   // two consecutive floats per thread
      const float* src = &xseq[((size_t)(bat0 + r) * L_DIM + (t + 1)) * I_DIM + k];
      __builtin_amdgcn_global_load_async_to_lds_b64(
          (AS1 v2i*)(void*)const_cast<float*>(src),
          (AS3 v2i*)(void*)&Xstage[tid * 2], 0, 0);
    }
#endif
    // Opaque copy of the weight base each step: stops LICM from hoisting all
    // 72 B-fragments (576 VGPRs) out of the t-loop and spilling to scratch.
    unsigned long long wpbits = (unsigned long long)Wp;
    asm volatile("" : "+s"(wpbits));
    // single shared base (SGPR) + single thread voffset; (kc,g) are immediates
    const AS1 char* Wb = (const AS1 char*)wpbits + voff;

    v8f acc[2][4];
    #pragma unroll
    for (int rt = 0; rt < 2; ++rt)
      #pragma unroll
      for (int g = 0; g < 4; ++g) acc[rt][g] = (v8f){0, 0, 0, 0, 0, 0, 0, 0};

    #define LOADB(kc, g) (*(const AS1 v16bf*)(Wb + ((kc) * 131072 + (g) * 32768)))

    v16bf bcur0 = LOADB(0, 0), bcur1 = LOADB(0, 1),
          bcur2 = LOADB(0, 2), bcur3 = LOADB(0, 3);
    for (int kc = 0; kc < KC_LSTM; ++kc) {
      v16bf a0 = *(const v16bf*)&Abuf[(0 * KC_LSTM + kc) * 512 + lane * 16];
      v16bf a1 = *(const v16bf*)&Abuf[(1 * KC_LSTM + kc) * 512 + lane * 16];
      v16bf bn0, bn1, bn2, bn3;
      if (kc + 1 < KC_LSTM) {     // issue next k-chunk's loads before the MACs
        bn0 = LOADB(kc + 1, 0); bn1 = LOADB(kc + 1, 1);
        bn2 = LOADB(kc + 1, 2); bn3 = LOADB(kc + 1, 3);
        gprefetch((const AS1 void*)(Wb + (kc + 1) * 131072));
      }
      acc[0][0] = wmma_bf16(a0, bcur0, acc[0][0]);
      acc[1][0] = wmma_bf16(a1, bcur0, acc[1][0]);
      acc[0][1] = wmma_bf16(a0, bcur1, acc[0][1]);
      acc[1][1] = wmma_bf16(a1, bcur1, acc[1][1]);
      acc[0][2] = wmma_bf16(a0, bcur2, acc[0][2]);
      acc[1][2] = wmma_bf16(a1, bcur2, acc[1][2]);
      acc[0][3] = wmma_bf16(a0, bcur3, acc[0][3]);
      acc[1][3] = wmma_bf16(a1, bcur3, acc[1][3]);
      if (kc + 1 < KC_LSTM) { bcur0 = bn0; bcur1 = bn1; bcur2 = bn2; bcur3 = bn3; }
    }
    #undef LOADB

    #pragma unroll
    for (int rt = 0; rt < 2; ++rt) {
      #pragma unroll
      for (int v = 0; v < 8; ++v) {
        float ig = sigm (acc[rt][0][v] + bias_i);
        float fg = sigm (acc[rt][1][v] + bias_f);
        float gg = tanhf(acc[rt][2][v] + bias_g);
        float og = sigm (acc[rt][3][v] + bias_o);
        float c  = fg * creg[rt][v] + ig * gg;
        creg[rt][v] = c;
        hnew[rt][v] = og * tanhf(c);
      }
    }
#if HAS_ASYNC
    if (t + 1 < L_DIM) __builtin_amdgcn_s_wait_asynccnt(0);
#endif
    __syncthreads();   // all Abuf reads of step t done; async LDS data visible
    // write h_t back into packed A (K = 64 + j)
    #pragma unroll
    for (int rt = 0; rt < 2; ++rt) {
      #pragma unroll
      for (int v = 0; v < 8; ++v) {
        int M = (lane >> 4) * 8 + v;
        int K = I_DIM + jcol, kc = K >> 5, ko = K & 31;
        Abuf[(rt * KC_LSTM + kc) * 512 + a_lane(ko, M) * 16 + a_elem(ko)] = (bf16)hnew[rt][v];
      }
    }
    if (t + 1 < L_DIM) {   // repack x_{t+1} (disjoint kc 0..1 region)
      for (int q = tid; q < 32 * I_DIM; q += 1024) {
#if HAS_ASYNC
        float x = Xstage[q];
#else
        int rr = q >> 6, kk = q & 63;
        float x = xseq[((size_t)(bat0 + rr) * L_DIM + (t + 1)) * I_DIM + kk];
#endif
        int r = q >> 6, k = q & 63;
        int kc = k >> 5, ko = k & 31, rt = r >> 4, M = r & 15;
        Abuf[(rt * KC_LSTM + kc) * 512 + a_lane(ko, M) * 16 + a_elem(ko)] = (bf16)x;
      }
    }
    __syncthreads();
  }
  // export final hidden, already in SDE A-frag layout (K 64..575 -> kc 0..15)
  for (int idx = tid; idx < 2 * 16 * 512; idx += 1024) {
    int rt = idx >> 13, rem = idx & 8191, kcs = rem >> 9, i = rem & 511;
    Hp[(size_t)blockIdx.x * (2 * 16 * 512) + idx] =
        Abuf[(rt * KC_LSTM + kcs + 2) * 512 + i];
  }
}

// ---------------------------------------------------------------------------
// Kernel 3: SDE. 64 WGs x 512 threads (16 waves). WG owns 32 batch rows.
// ---------------------------------------------------------------------------
__global__ __launch_bounds__(512) void sde_kernel(
    const float* __restrict__ ts, const float* __restrict__ y0,
    const bf16* __restrict__ Wsp, const float* __restrict__ bias2,
    const bf16* __restrict__ Hp, float* __restrict__ out) {
  __shared__ __align__(128) bf16 Abuf[2 * KC_SDE * 512];  // 36864 B
  __shared__ float Gbuf[32 * N_SDE];                      // 16 KB drift|diff
  __shared__ float yS[32 * O_DIM];                        // 8 KB state
  const int tid  = threadIdx.x;
  const int wave = tid >> 5;       // 0..15
  const int lane = tid & 31;
  const int bat0 = blockIdx.x * 32;
  const int rt   = wave >> 3;      // row tile 0..1
  const int ct   = wave & 7;       // output col tile 0..7
  const float bn = bias2[ct * 16 + (lane & 15)];          // time-invariant bias
  // fragment (kc) for this wave lives at  base + voff + kc*8192
  const AS1 char* Wsb = (const AS1 char*)(unsigned long long)Wsp
                      + (unsigned)(ct * 1024 + lane * 32);

  for (int idx = tid; idx < 2 * 16 * 512; idx += 512) {   // hid -> kc 0..15
    int rt2 = idx >> 13, rem = idx & 8191, kcs = rem >> 9, i = rem & 511;
    Abuf[(rt2 * KC_SDE + kcs) * 512 + i] = Hp[(size_t)blockIdx.x * (2 * 16 * 512) + idx];
  }
  for (int q = tid; q < 32 * O_DIM; q += 512) {           // y0 -> state/out/A
    int r = q >> 6, j = q & 63;
    float y = y0[(size_t)(bat0 + r) * O_DIM + j];
    yS[q] = y;
    out[(size_t)(bat0 + r) * O_DIM + j] = y;              // t = 0 plane
    int K = H_DIM + j, kc = K >> 5, ko = K & 31, rr = r >> 4, M = r & 15;
    Abuf[(rr * KC_SDE + kc) * 512 + a_lane(ko, M) * 16 + a_elem(ko)] = (bf16)y;
  }
  __syncthreads();

  const float dt  = ts[1] - ts[0];
  const float sdt = __builtin_sqrtf(dt);
  const unsigned jthr = (unsigned)(5e-8 * 4294967296.0);  // JUMP_PROB * 2^32

  for (int t = 0; t < T_TOT - 1; ++t) {
    v8f acc = {0, 0, 0, 0, 0, 0, 0, 0};
    for (int kc = 0; kc < KC_SDE; ++kc) {
      v16bf a = *(const v16bf*)&Abuf[(rt * KC_SDE + kc) * 512 + lane * 16];
      v16bf b = *(const AS1 v16bf*)(Wsb + kc * 8192);
      acc = wmma_bf16(a, b, acc);
    }
    #pragma unroll
    for (int v = 0; v < 8; ++v) {
      int r = rt * 16 + (lane >> 4) * 8 + v;
      int n = ct * 16 + (lane & 15);
      Gbuf[r * N_SDE + n] = acc[v] + bn;
    }
    __syncthreads();
    for (int q = tid; q < 32 * O_DIM; q += 512) {
      int r = q >> 6, j = q & 63;
      float dr = Gbuf[r * N_SDE + j];
      float df = Gbuf[r * N_SDE + O_DIM + j];
      unsigned ge = (unsigned)((bat0 + r) * O_DIM + j);
      uint2 r1 = tf2x32(0x2Au, 0x9E3779B9u ^ (unsigned)t, ge, (unsigned)t * 3u);
      float bm = bm_normal(r1.x, r1.y) * sdt;
      uint2 r2 = tf2x32(0x2Au ^ 0x632BE59Bu, 0x9E3779B9u ^ (unsigned)t, ge,
                        (unsigned)t * 3u + 1u);
      float jump = (r2.x < jthr) ? bm_normal(r2.y, r2.x) * 0.05f : 0.f;
      float y = yS[q] + dr * dt + df * bm + jump;
      yS[q] = y;
      out[(size_t)(t + 1) * (B_DIM * O_DIM) + (size_t)(bat0 + r) * O_DIM + j] = y;
      int K = H_DIM + j, kc = K >> 5, ko = K & 31, rr = r >> 4, M = r & 15;
      Abuf[(rr * KC_SDE + kc) * 512 + a_lane(ko, M) * 16 + a_elem(ko)] = (bf16)y;
    }
    __syncthreads();
  }
}

// ---------------------------------------------------------------------------
extern "C" void kernel_launch(void* const* d_in, const int* in_sizes, int n_in,
                              void* d_out, int out_size, void* d_ws, size_t ws_size,
                              hipStream_t stream) {
  (void)in_sizes; (void)n_in; (void)out_size; (void)ws_size;
  const float* ts      = (const float*)d_in[0];
  const float* y0      = (const float*)d_in[1];
  const float* xseq    = (const float*)d_in[2];
  const float* W_ih    = (const float*)d_in[3];
  const float* W_hh    = (const float*)d_in[4];
  const float* b_ih    = (const float*)d_in[5];
  const float* b_hh    = (const float*)d_in[6];
  const float* drift_W = (const float*)d_in[7];
  const float* drift_b = (const float*)d_in[8];
  const float* diff_W  = (const float*)d_in[9];
  const float* diff_b  = (const float*)d_in[10];
  float* out = (float*)d_out;

  // workspace carve (all offsets 256B-aligned)
  char* w = (char*)d_ws;
  bf16*  Wp    = (bf16*)w;  w += (size_t)KC_LSTM * CT_LSTM * 32 * 16 * 2; // 2359296
  bf16*  Wsp   = (bf16*)w;  w += (size_t)KC_SDE  * CT_SDE  * 32 * 16 * 2; //  147456
  float* biasG = (float*)w; w += (size_t)N_GATE * 4;                      //    8192
  float* bias2 = (float*)w; w += (size_t)N_SDE * 4;                       //     512
  bf16*  Hp    = (bf16*)w;                                                // 2097152

  const int packN = KC_LSTM * CT_LSTM * 32 * 16 + KC_SDE * CT_SDE * 32 * 16
                  + N_GATE + N_SDE;
  pack_kernel<<<(packN + 255) / 256, 256, 0, stream>>>(
      W_ih, W_hh, b_ih, b_hh, drift_W, drift_b, diff_W, diff_b,
      Wp, Wsp, biasG, bias2);
  lstm_kernel<<<B_DIM / 32, 1024, 0, stream>>>(xseq, Wp, biasG, Hp);
  sde_kernel<<<B_DIM / 32, 512, 0, stream>>>(ts, y0, Wsp, bias2, Hp, out);
}
